// GCN_31241592111373
// MI455X (gfx1250) — matrix operand order repaired and checked by reading, not compiled, not run on hardware
//
#include <hip/hip_runtime.h>
#include <math.h>

// ---- CDNA5 WMMA vector types (probe-confirmed signatures) ----
typedef __attribute__((ext_vector_type(16))) __bf16 v16bf;
typedef __attribute__((ext_vector_type(8)))  __bf16 v8bf;
typedef __attribute__((ext_vector_type(8)))  float  v8f;

#define CH   64   // IN_C == HID_C
#define OUTC 16

// ---------------- elementwise helpers ----------------

__global__ void k_cvt_bf16(const float* __restrict__ in, __bf16* __restrict__ out, int n) {
    int i = blockIdx.x * blockDim.x + threadIdx.x;
    if (i < n) out[i] = (__bf16)in[i];
}

__global__ void k_deg_init(float* __restrict__ deg, int n) {
    int i = blockIdx.x * blockDim.x + threadIdx.x;
    if (i < n) deg[i] = 1.0f;   // self-loop contributes 1
}

__global__ void k_deg_count(const int* __restrict__ dst, float* __restrict__ deg, int e) {
    int i = blockIdx.x * blockDim.x + threadIdx.x;
    if (i < e) atomicAdd(&deg[dst[i]], 1.0f);
}

__global__ void k_dinv(float* __restrict__ deg, int n) {
    int i = blockIdx.x * blockDim.x + threadIdx.x;
    if (i < n) deg[i] = rsqrtf(fmaxf(deg[i], 1.0f));  // deg >= 1 always (self loops)
}

// agg[i][c] = dinv[i]^2 * lin[i][c]   (self-loop message; also fully initializes agg)
__global__ void k_agg_init(const float* __restrict__ dinv, const float* __restrict__ lin,
                           float* __restrict__ agg, int total) {
    int i = blockIdx.x * blockDim.x + threadIdx.x;
    if (i < total) {
        float w = dinv[i >> 6];
        agg[i] = w * w * lin[i];
    }
}

// one wave per edge; lane handles 2 channels (float2). L2-resident f32 atomics.
__global__ void k_edge_scatter(const int* __restrict__ src, const int* __restrict__ dst,
                               const float* __restrict__ dinv, const float* __restrict__ lin,
                               float* __restrict__ agg, int e) {
    int gid  = blockIdx.x * blockDim.x + threadIdx.x;
    int edge = gid >> 5;
    int lane = threadIdx.x & 31;
    if (edge >= e) return;
    int s = src[edge];
    int d = dst[edge];
    float w = dinv[s] * dinv[d];
    const float2 v = *(const float2*)(lin + (size_t)s * CH + lane * 2);
    float* dp = agg + (size_t)d * CH + lane * 2;
    atomicAdd(dp + 0, w * v.x);
    atomicAdd(dp + 1, w * v.y);
}

__global__ void k_bias_relu_cvt(const float* __restrict__ agg, const float* __restrict__ bias,
                                __bf16* __restrict__ out, int total) {
    int i = blockIdx.x * blockDim.x + threadIdx.x;
    if (i < total) {
        float v = agg[i] + bias[i & (CH - 1)];
        out[i] = (__bf16)fmaxf(v, 0.0f);
    }
}

__global__ void k_logsoftmax(const float* __restrict__ logits, float* __restrict__ out, int n) {
    int i = blockIdx.x * blockDim.x + threadIdx.x;
    if (i >= n) return;
    const float* p = logits + (size_t)i * OUTC;
    float m = -INFINITY;
#pragma unroll
    for (int c = 0; c < OUTC; ++c) m = fmaxf(m, p[c]);
    float s = 0.0f;
#pragma unroll
    for (int c = 0; c < OUTC; ++c) s += expf(p[c] - m);
    float l = logf(s);
    float* q = out + (size_t)i * OUTC;
#pragma unroll
    for (int c = 0; c < OUTC; ++c) q[c] = p[c] - m - l;
}

// ---------------- B-matrix pre-swizzle into WMMA fragment layout (fused f32->bf16) ----------------
// Packed layout: Bp[((kstep*NCT + ct)*32 + lane)*16 + e]
//   = B[K = kstep*32 + (e<8 ? (lane>>4)*8 + e : 16 + (lane>>4)*8 + (e-8))][col = ct*16 + (lane&15)]
// so each lane's 16 fragment elements are contiguous (one 32B load in the GEMM).
template <int NCOLS>
__global__ void k_pack_b(const float* __restrict__ W, __bf16* __restrict__ Bp) {
    constexpr int NCT = NCOLS / 16;
    int idx = blockIdx.x * blockDim.x + threadIdx.x;   // total CH*NCOLS elements
    if (idx >= CH * NCOLS) return;
    int e     = idx & 15;
    int lane  = (idx >> 4) & 31;
    int frag  = idx >> 9;           // 512 elements per fragment block
    int ct    = frag % NCT;
    int kstep = frag / NCT;
    int K   = kstep * 32 + (e < 8 ? (lane >> 4) * 8 + e : 16 + (lane >> 4) * 8 + (e - 8));
    int col = ct * 16 + (lane & 15);
    Bp[idx] = (__bf16)W[K * NCOLS + col];
}

// ---------------- bf16 WMMA GEMM: C[nrows x NCOLS] = A[nrows x 64] * Bp (+bias) ----------------
// One wave computes one 16-row tile across all NCOLS/16 column tiles.
// A fragment (16-bit A, 16x32): lane = row (m = lane&15); half-wave hf = lane>>4 selects
//   elements 0..7  -> K = ks*32 + hf*8 + e        (contiguous 16B load)
//   elements 8..15 -> K = ks*32 + 16 + hf*8 + e   (contiguous 16B load)
// B fragments come pre-swizzled: one contiguous 32B load per lane.
// D tile (f32 16x16): element r -> row r + hf*8, column lane%16
template <int NCOLS>
__global__ void k_gemm_bf16(const __bf16* __restrict__ A, const __bf16* __restrict__ Bp,
                            float* __restrict__ C, const float* __restrict__ bias, int nrows) {
    const int lane   = threadIdx.x & 31;
    const int wid    = threadIdx.x >> 5;
    const int tile   = blockIdx.x * (blockDim.x >> 5) + wid;
    const int ntiles = nrows >> 4;              // nrows is a multiple of 16 here
    if (tile >= ntiles) return;                 // wave-uniform: EXEC stays all-ones for WMMA

    const int m    = lane & 15;
    const int hf   = lane >> 4;
    const int row0 = tile << 4;
    constexpr int NCT = NCOLS / 16;

    v8f acc[NCT];
#pragma unroll
    for (int ct = 0; ct < NCT; ++ct) acc[ct] = {};

#pragma unroll
    for (int ks = 0; ks < 2; ++ks) {
        const __bf16* ap = A + (size_t)(row0 + m) * CH + ks * 32 + hf * 8;
        v8bf alo = *(const v8bf*)ap;
        v8bf ahi = *(const v8bf*)(ap + 16);
        v16bf a;
#pragma unroll
        for (int i = 0; i < 8; ++i) { a[i] = alo[i]; a[8 + i] = ahi[i]; }

#pragma unroll
        for (int ct = 0; ct < NCT; ++ct) {
            v16bf b = *(const v16bf*)(Bp + (((ks * NCT + ct) << 5) + lane) * 16);
            acc[ct] = __builtin_amdgcn_wmma_f32_16x16x32_bf16(
                false, a, false, b, (short)0, acc[ct], false, false);
        }
    }

#pragma unroll
    for (int ct = 0; ct < NCT; ++ct) {
        const int col = (ct << 4) + m;
        const float bv = bias ? bias[col] : 0.0f;
#pragma unroll
        for (int r = 0; r < 8; ++r) {
            const int row = row0 + r + (hf << 3);
            C[(size_t)row * NCOLS + col] = acc[ct][r] + bv;
        }
    }
}

// ---------------- launcher ----------------

extern "C" void kernel_launch(void* const* d_in, const int* in_sizes, int n_in,
                              void* d_out, int out_size, void* d_ws, size_t ws_size,
                              hipStream_t stream) {
    (void)n_in; (void)out_size; (void)ws_size;

    const float* x   = (const float*)d_in[0];
    const int*   ei  = (const int*)d_in[1];
    const float* W1  = (const float*)d_in[2];
    const float* b1  = (const float*)d_in[3];
    const float* W2  = (const float*)d_in[4];
    const float* b2  = (const float*)d_in[5];
    const float* Wfc = (const float*)d_in[6];
    const float* bfc = (const float*)d_in[7];
    float* out = (float*)d_out;

    const int n = in_sizes[0] / CH;   // 100000 nodes (multiple of 16)
    const int e = in_sizes[1] / 2;    // 1600000 edges
    const int* src = ei;
    const int* dst = ei + e;

    // workspace carve-out (256B aligned slices)
    char*  ws  = (char*)d_ws;
    size_t off = 0;
    auto alloc = [&](size_t bytes) -> void* {
        void* p = ws + off;
        off = (off + bytes + 255) & ~(size_t)255;
        return p;
    };
    float*  deg    = (float*)alloc((size_t)n * 4);            // reused as dinv
    __bf16* W1p    = (__bf16*)alloc((size_t)CH * CH * 2);     // pre-swizzled fragments
    __bf16* W2p    = (__bf16*)alloc((size_t)CH * CH * 2);
    __bf16* Wfcp   = (__bf16*)alloc((size_t)CH * OUTC * 2);
    __bf16* featb  = (__bf16*)alloc((size_t)n * CH * 2);      // current layer input (bf16)
    float*  lin    = (float*)alloc((size_t)n * CH * 4);       // GEMM output
    float*  agg    = (float*)alloc((size_t)n * CH * 4);       // scatter accumulator
    float*  logits = (float*)alloc((size_t)n * OUTC * 4);

    const int B256  = 256;
    const int total = n * CH;
    dim3 blk(B256);
    auto nb = [&](int cnt) { return dim3((cnt + B256 - 1) / B256); };

    // degrees -> dinv (in place)
    k_deg_init<<<nb(n), blk, 0, stream>>>(deg, n);
    k_deg_count<<<nb(e), blk, 0, stream>>>(dst, deg, e);
    k_dinv<<<nb(n), blk, 0, stream>>>(deg, n);

    // feature conversion + weight pack (fused cvt+swizzle)
    k_cvt_bf16<<<nb(total), blk, 0, stream>>>(x, featb, total);
    k_pack_b<CH><<<nb(CH * CH), blk, 0, stream>>>(W1, W1p);
    k_pack_b<CH><<<nb(CH * CH), blk, 0, stream>>>(W2, W2p);
    k_pack_b<OUTC><<<nb(CH * OUTC), blk, 0, stream>>>(Wfc, Wfcp);

    const int  ntiles = n >> 4;                      // 6250
    const dim3 gemm_blk(128);                        // 4 waves / block
    const dim3 gemm_grd((ntiles + 3) / 4);
    const dim3 edge_grd((e + 7) / 8);                // 1 wave per edge, 8 waves/block

    // ---- layer 1 ----
    k_gemm_bf16<CH><<<gemm_grd, gemm_blk, 0, stream>>>(featb, W1p, lin, nullptr, n);
    k_agg_init<<<nb(total), blk, 0, stream>>>(deg, lin, agg, total);
    k_edge_scatter<<<edge_grd, blk, 0, stream>>>(src, dst, deg, lin, agg, e);
    k_bias_relu_cvt<<<nb(total), blk, 0, stream>>>(agg, b1, featb, total);

    // ---- layer 2 ----
    k_gemm_bf16<CH><<<gemm_grd, gemm_blk, 0, stream>>>(featb, W2p, lin, nullptr, n);
    k_agg_init<<<nb(total), blk, 0, stream>>>(deg, lin, agg, total);
    k_edge_scatter<<<edge_grd, blk, 0, stream>>>(src, dst, deg, lin, agg, e);
    k_bias_relu_cvt<<<nb(total), blk, 0, stream>>>(agg, b2, featb, total);

    // ---- head + log_softmax ----
    k_gemm_bf16<OUTC><<<gemm_grd, gemm_blk, 0, stream>>>(featb, Wfcp, logits, bfc, n);
    k_logsoftmax<<<nb(n), blk, 0, stream>>>(logits, out, n);
}
